// double_deform_conv_41686952575680
// MI455X (gfx1250) — matrix operand order repaired and checked by reading, not compiled, not run on hardware
//
#include <hip/hip_runtime.h>
#include <stdint.h>

// Problem constants (B,Cin,Cout,H,W = 2,64,128,128,128)
#define BATCH    2
#define C_IN     64
#define C_OUT    128
#define HH       128
#define WW       128
#define HW_SHIFT 14            // H*W = 16384
#define HW_MASK  ((1 << HW_SHIFT) - 1)
#define NK       9             // 3x3 taps
#define K1       (NK * C_IN)           // 576
#define K2       (NK * C_OUT + C_IN)   // 1216 (conv2 + folded identity)
#define M_TOTAL  (BATCH << HW_SHIFT)   // 32768

typedef __attribute__((ext_vector_type(16))) __bf16        v16bf;
typedef __attribute__((ext_vector_type(8)))  float         v8f;
typedef __attribute__((ext_vector_type(4)))  unsigned int  u32x4;

union FragBF { u32x4 q[2]; v16bf v; };

__device__ __forceinline__ uint16_t f32_to_bf16(float f) {
  uint32_t u = __float_as_uint(f);
  uint32_t r = (u + 0x7FFFu + ((u >> 16) & 1u)) >> 16;  // round-to-nearest-even
  return (uint16_t)r;
}

// CDNA5 async global->LDS copy (ASYNCcnt-tracked, bypasses VGPRs).
// vdst = per-lane LDS byte address, vaddr = per-lane 64-bit global address.
__device__ __forceinline__ void async_copy_b128(uint32_t lds_addr, const void* gptr) {
  asm volatile("global_load_async_to_lds_b128 %0, %1, off"
               :
               : "v"(lds_addr), "v"((uint64_t)(uintptr_t)gptr)
               : "memory");
}
__device__ __forceinline__ void wait_asynccnt0() {
  asm volatile("s_wait_asynccnt 0x0" ::: "memory");
}

// ---------------------------------------------------------------------------
// BN coefficient prep: s=g/sqrt(v+eps), t=b-m*s; bias2 = t2 + t3 + s3*id_b
// ---------------------------------------------------------------------------
__global__ void bn_coeffs_kernel(const float* g1, const float* b1, const float* m1, const float* v1,
                                 const float* g2, const float* b2, const float* m2, const float* v2,
                                 const float* g3, const float* b3, const float* m3, const float* v3,
                                 const float* id_b,
                                 float* s1, float* t1, float* s2, float* s3, float* bias2) {
  int o = threadIdx.x;
  if (o >= C_OUT) return;
  float ss1 = g1[o] * rsqrtf(v1[o] + 1e-5f);
  s1[o] = ss1;
  t1[o] = b1[o] - m1[o] * ss1;
  float ss2 = g2[o] * rsqrtf(v2[o] + 1e-5f);
  float tt2 = b2[o] - m2[o] * ss2;
  float ss3 = g3[o] * rsqrtf(v3[o] + 1e-5f);
  float tt3 = b3[o] - m3[o] * ss3;
  s2[o] = ss2;
  s3[o] = ss3;
  bias2[o] = tt2 + tt3 + ss3 * id_b[o];
}

// ---------------------------------------------------------------------------
// Pack conv1 weights (Cout,Cin,3,3) -> bf16 Wt1[o][k], k = n*C_IN + c
// ---------------------------------------------------------------------------
__global__ void pack_w1_kernel(const float* __restrict__ w, uint16_t* __restrict__ out) {
  int tid = blockIdx.x * blockDim.x + threadIdx.x;
  if (tid >= C_OUT * K1) return;
  int o = tid / K1, k = tid % K1;
  int n = k / C_IN, c = k % C_IN;
  out[tid] = f32_to_bf16(w[(o * C_IN + c) * NK + n]);
}

// ---------------------------------------------------------------------------
// Pack conv2 weights scaled by s2 (k<9*C_OUT) and identity weights scaled by
// s3 (k>=9*C_OUT) -> bf16 Wt2[o][k], K2 = 1216
// ---------------------------------------------------------------------------
__global__ void pack_w2_kernel(const float* __restrict__ w2, const float* __restrict__ idw,
                               const float* __restrict__ s2, const float* __restrict__ s3,
                               uint16_t* __restrict__ out) {
  int tid = blockIdx.x * blockDim.x + threadIdx.x;
  if (tid >= C_OUT * K2) return;
  int o = tid / K2, k = tid % K2;
  float val;
  if (k < NK * C_OUT) {
    int n = k / C_OUT, c = k % C_OUT;
    val = s2[o] * w2[(o * C_OUT + c) * NK + n];
  } else {
    int c = k - NK * C_OUT;
    val = s3[o] * idw[o * C_IN + c];
  }
  out[tid] = f32_to_bf16(val);
}

// ---------------------------------------------------------------------------
// Offset conv: 3x3, pad 1, 18 output channels.  One thread per (b,ch,h,w).
// Grid total = BATCH*18*H*W (exact multiple of 256).
// ---------------------------------------------------------------------------
__global__ void offset_conv_kernel(const float* __restrict__ x, const float* __restrict__ ow,
                                   const float* __restrict__ ob, float* __restrict__ off, int C) {
  int tid = blockIdx.x * blockDim.x + threadIdx.x;
  int w  = tid & (WW - 1);
  int h  = (tid >> 7) & (HH - 1);
  int ch = (tid >> HW_SHIFT) % 18;
  int b  = tid / (18 << HW_SHIFT);
  float acc = ob[ch];
  const float* xb = x + ((size_t)b * C << HW_SHIFT);
  for (int ky = 0; ky < 3; ++ky) {
    int iy = h + ky - 1;
    if (iy < 0 || iy >= HH) continue;
    for (int kx = 0; kx < 3; ++kx) {
      int ix = w + kx - 1;
      if (ix < 0 || ix >= WW) continue;
      const float* xp = xb + (iy << 7) + ix;
      const float* wp = ow + (size_t)ch * C * NK + ky * 3 + kx;
      for (int c = 0; c < C; ++c)
        acc += wp[c * NK] * xp[(size_t)c << HW_SHIFT];
    }
  }
  off[tid] = acc;  // layout (B,18,H,W) flat == tid
}

// ---------------------------------------------------------------------------
// Deformable bilinear sampling -> bf16 im2col.  One thread per (b,n,h,w),
// inner loop over channels.  col[p][n*C + c], p = (b,h,w) pixel index.
// ---------------------------------------------------------------------------
__global__ void deform_sample_kernel(const float* __restrict__ x, const float* __restrict__ off,
                                     uint16_t* __restrict__ col, int C, int Kstride) {
  int tid = blockIdx.x * blockDim.x + threadIdx.x;
  int w = tid & (WW - 1);
  int h = (tid >> 7) & (HH - 1);
  int n = (tid >> HW_SHIFT) % NK;
  int b = tid / (NK << HW_SHIFT);
  int hw = (h << 7) + w;

  float offy = off[(((size_t)b * 18 + n) << HW_SHIFT) + hw];
  float offx = off[(((size_t)b * 18 + 9 + n) << HW_SHIFT) + hw];
  float py = offy + (float)(n / 3 - 1) + (float)(h + 1);
  float px = offx + (float)(n % 3 - 1) + (float)(w + 1);
  py = fminf(fmaxf(py, 0.0f), (float)(HH + 1));   // [0, Hp-1]
  px = fminf(fmaxf(px, 0.0f), (float)(WW + 1));
  float fy = floorf(py), fx = floorf(px);
  int iy0 = (int)fy, ix0 = (int)fx;
  int iy1 = min(iy0 + 1, HH + 1), ix1 = min(ix0 + 1, WW + 1);
  float wy0 = 1.0f + (fy - py);
  float wy1 = 1.0f - ((float)iy1 - py);
  float wx0 = 1.0f + (fx - px);
  float wx1 = 1.0f - ((float)ix1 - px);
  // zero-pad ring: real pixels are iy in [1,H], ix in [1,W]
  bool y0v = (iy0 >= 1 && iy0 <= HH), y1v = (iy1 >= 1 && iy1 <= HH);
  bool x0v = (ix0 >= 1 && ix0 <= WW), x1v = (ix1 >= 1 && ix1 <= WW);
  float w00 = (y0v && x0v) ? wy0 * wx0 : 0.0f;
  float w01 = (y0v && x1v) ? wy0 * wx1 : 0.0f;
  float w10 = (y1v && x0v) ? wy1 * wx0 : 0.0f;
  float w11 = (y1v && x1v) ? wy1 * wx1 : 0.0f;
  int o00 = (y0v && x0v) ? ((iy0 - 1) << 7) + (ix0 - 1) : 0;
  int o01 = (y0v && x1v) ? ((iy0 - 1) << 7) + (ix1 - 1) : 0;
  int o10 = (y1v && x0v) ? ((iy1 - 1) << 7) + (ix0 - 1) : 0;
  int o11 = (y1v && x1v) ? ((iy1 - 1) << 7) + (ix1 - 1) : 0;

  const float* xb = x + ((size_t)b * C << HW_SHIFT);
  size_t p = ((size_t)b << HW_SHIFT) + hw;
  uint16_t* cp = col + p * (size_t)Kstride + n * C;
  for (int c = 0; c < C; ++c) {
    const float* base = xb + ((size_t)c << HW_SHIFT);
    float v = w00 * base[o00] + w01 * base[o01] + w10 * base[o10] + w11 * base[o11];
    cp[c] = f32_to_bf16(v);
  }
}

// ---------------------------------------------------------------------------
// Append bf16(x) columns for the folded identity path: col2[p][1152 + c]
// ---------------------------------------------------------------------------
__global__ void append_x_kernel(const float* __restrict__ x, uint16_t* __restrict__ col) {
  int tid = blockIdx.x * blockDim.x + threadIdx.x;
  int c = tid & (C_IN - 1);
  int p = tid >> 6;
  int b = p >> HW_SHIFT;
  float v = x[(((size_t)b * C_IN + c) << HW_SHIFT) + (p & HW_MASK)];
  col[(size_t)p * K2 + NK * C_OUT + c] = f32_to_bf16(v);
}

// ---------------------------------------------------------------------------
// bf16 WMMA GEMM: out[p][o] = relu( (A[p][:] . Wt[o][:]) * s?[o] + t[o] )
// A: M x K bf16 row-major (K % 64 == 0); Wt: N x K bf16 (N-major, contiguous K)
// Workgroup: 256 threads = 8 waves, 128x128 tile; wave = 2x4 WMMA tiles.
// A-tiles double-buffered in LDS via async global->LDS copies (ASYNCcnt);
// K-loop unrolled 2x so the ping-pong parity is static.
// Output scattered to NCHW fp32.
// ---------------------------------------------------------------------------
__global__ __launch_bounds__(256) void gemm_bf16_wmma_kernel(
    const uint16_t* __restrict__ Amat, const uint16_t* __restrict__ Wt,
    float* __restrict__ out, int K, const float* __restrict__ s,
    const float* __restrict__ t, int use_scale) {
  __shared__ uint16_t sA[2][128 * 32];   // 2 x (128 rows x 32 K-halves), 16 KB

  const int tid     = threadIdx.x;
  const int lane    = tid & 31;
  const int waveId  = tid >> 5;
  const int wm      = waveId & 3;     // wave grid 4(M) x 2(N)
  const int wn      = waveId >> 2;
  const int m_block = blockIdx.x * 128;
  const int lrow    = lane & 15;
  const int lhi     = lane >> 4;      // 0 or 1 (half-wave select)

  v8f acc[2][4];
  const v8f vzero = {0.f, 0.f, 0.f, 0.f, 0.f, 0.f, 0.f, 0.f};
#pragma unroll
  for (int i = 0; i < 2; ++i)
#pragma unroll
    for (int j = 0; j < 4; ++j) acc[i][j] = vzero;

  const int ld_row  = tid >> 1;        // 0..127
  const int ld_half = (tid & 1) << 4;  // 0 or 16 halves (32B per thread)

  // LDS aperture: low 32 bits of generic address are the LDS byte offset.
  const uint32_t lds0 = (uint32_t)(uintptr_t)(&sA[0][ld_row * 32 + ld_half]);
  const uint32_t lds1 = (uint32_t)(uintptr_t)(&sA[1][ld_row * 32 + ld_half]);
  const uint16_t* srcBase = Amat + (size_t)(m_block + ld_row) * K + ld_half;

  // prologue: stage first tile into buffer 0
  async_copy_b128(lds0,      srcBase);
  async_copy_b128(lds0 + 16, srcBase + 8);

#pragma unroll 2
  for (int k0 = 0; k0 < K; k0 += 32) {
    const int cur = (k0 >> 5) & 1;   // static after 2x unroll (K/32 is even)
    wait_asynccnt0();     // my portion of buf[cur] landed in LDS
    __syncthreads();      // everyone's portion landed; buf[cur^1] reads retired
    if (k0 + 32 < K) {    // kick off next tile into the other buffer
      const uint16_t* nsrc = srcBase + k0 + 32;
      const uint32_t  nl   = cur ? lds0 : lds1;
      async_copy_b128(nl,      nsrc);
      async_copy_b128(nl + 16, nsrc + 8);
    }

    // B fragments straight from global (weights are L2/WGP$-resident):
    // lanes 0-15 hold K=k0..k0+15 of column n, lanes 16-31 hold K=k0+16..k0+31
    FragBF bf[4];
#pragma unroll
    for (int j = 0; j < 4; ++j) {
      const uint16_t* wp = Wt + (size_t)(wn * 64 + j * 16 + lrow) * K + k0 + lhi * 16;
      bf[j].q[0] = *(const u32x4*)(wp);
      bf[j].q[1] = *(const u32x4*)(wp + 8);
    }

    // A fragments in contiguous K order (same as B):
    // lanes 0-15: K {0..15}; lanes 16-31: K {16..31}
#pragma unroll
    for (int i = 0; i < 2; ++i) {
      FragBF af;
      const uint16_t* ap = &sA[cur][(wm * 32 + i * 16 + lrow) * 32 + lhi * 16];
      af.q[0] = *(const u32x4*)(ap);
      af.q[1] = *(const u32x4*)(ap + 8);
#pragma unroll
      for (int j = 0; j < 4; ++j)
        acc[i][j] = __builtin_amdgcn_wmma_f32_16x16x32_bf16(
            false, af.v, false, bf[j].v, (short)0, acc[i][j], false, false);
    }
  }

  // Epilogue: C/D layout -> VGPR r, lanes0-15: (M=r, N=lane); lanes16-31: (M=r+8)
#pragma unroll
  for (int i = 0; i < 2; ++i) {
#pragma unroll
    for (int j = 0; j < 4; ++j) {
      int o = wn * 64 + j * 16 + lrow;
      float ss = use_scale ? s[o] : 1.0f;
      float tt = t[o];
#pragma unroll
      for (int r = 0; r < 8; ++r) {
        int p = m_block + wm * 32 + i * 16 + lhi * 8 + r;
        float v = fmaxf(acc[i][j][r] * ss + tt, 0.0f);
        int b = p >> HW_SHIFT;
        out[(((size_t)(b * C_OUT + o)) << HW_SHIFT) + (p & HW_MASK)] = v;
      }
    }
  }
}

// ---------------------------------------------------------------------------
// Host driver
// ---------------------------------------------------------------------------
extern "C" void kernel_launch(void* const* d_in, const int* in_sizes, int n_in,
                              void* d_out, int out_size, void* d_ws, size_t ws_size,
                              hipStream_t stream) {
  const float* x        = (const float*)d_in[0];
  const float* dc1_offw = (const float*)d_in[1];
  const float* dc1_offb = (const float*)d_in[2];
  const float* dc1_w    = (const float*)d_in[3];
  const float* bn1_g    = (const float*)d_in[4];
  const float* bn1_b    = (const float*)d_in[5];
  const float* bn1_m    = (const float*)d_in[6];
  const float* bn1_v    = (const float*)d_in[7];
  const float* dc2_offw = (const float*)d_in[8];
  const float* dc2_offb = (const float*)d_in[9];
  const float* dc2_w    = (const float*)d_in[10];
  const float* bn2_g    = (const float*)d_in[11];
  const float* bn2_b    = (const float*)d_in[12];
  const float* bn2_m    = (const float*)d_in[13];
  const float* bn2_v    = (const float*)d_in[14];
  const float* id_w     = (const float*)d_in[15];
  const float* id_b     = (const float*)d_in[16];
  const float* bn3_g    = (const float*)d_in[17];
  const float* bn3_b    = (const float*)d_in[18];
  const float* bn3_m    = (const float*)d_in[19];
  const float* bn3_v    = (const float*)d_in[20];
  float* out = (float*)d_out;

  char* ws = (char*)d_ws;
  size_t cur = 0;
  auto alloc = [&](size_t bytes) -> void* {
    cur = (cur + 255) & ~(size_t)255;
    void* p = ws + cur;
    cur += bytes;
    return p;
  };
  float*    s1    = (float*)alloc(C_OUT * sizeof(float));
  float*    t1    = (float*)alloc(C_OUT * sizeof(float));
  float*    s2    = (float*)alloc(C_OUT * sizeof(float));
  float*    s3    = (float*)alloc(C_OUT * sizeof(float));
  float*    bias2 = (float*)alloc(C_OUT * sizeof(float));
  uint16_t* Wt1   = (uint16_t*)alloc((size_t)C_OUT * K1 * sizeof(uint16_t));
  uint16_t* Wt2   = (uint16_t*)alloc((size_t)C_OUT * K2 * sizeof(uint16_t));
  float*    off1  = (float*)alloc((size_t)BATCH * 18 * (1 << HW_SHIFT) * sizeof(float));
  float*    off2  = (float*)alloc((size_t)BATCH * 18 * (1 << HW_SHIFT) * sizeof(float));
  float*    h1    = (float*)alloc((size_t)BATCH * C_OUT * (1 << HW_SHIFT) * sizeof(float));
  uint16_t* col1  = (uint16_t*)alloc((size_t)M_TOTAL * K1 * sizeof(uint16_t));
  uint16_t* col2  = (uint16_t*)alloc((size_t)M_TOTAL * K2 * sizeof(uint16_t));
  (void)ws_size; (void)in_sizes; (void)n_in; (void)out_size;

  // 1) BN coefficients
  bn_coeffs_kernel<<<1, 128, 0, stream>>>(bn1_g, bn1_b, bn1_m, bn1_v,
                                          bn2_g, bn2_b, bn2_m, bn2_v,
                                          bn3_g, bn3_b, bn3_m, bn3_v,
                                          id_b, s1, t1, s2, s3, bias2);
  // 2) Weight packing (bf16, N-major)
  pack_w1_kernel<<<(C_OUT * K1 + 255) / 256, 256, 0, stream>>>(dc1_w, Wt1);
  pack_w2_kernel<<<(C_OUT * K2 + 255) / 256, 256, 0, stream>>>(dc2_w, id_w, s2, s3, Wt2);

  // 3) Layer 1: offsets -> sample -> GEMM(+BN1+ReLU) -> h1 (NCHW fp32)
  offset_conv_kernel<<<(BATCH * 18 << HW_SHIFT) / 256, 256, 0, stream>>>(x, dc1_offw, dc1_offb, off1, C_IN);
  deform_sample_kernel<<<(BATCH * NK << HW_SHIFT) / 256, 256, 0, stream>>>(x, off1, col1, C_IN, K1);
  gemm_bf16_wmma_kernel<<<M_TOTAL / 128, 256, 0, stream>>>(col1, Wt1, h1, K1, s1, t1, 1);

  // 4) Layer 2: offsets -> sample -> append identity columns -> GEMM(+bias+ReLU)
  offset_conv_kernel<<<(BATCH * 18 << HW_SHIFT) / 256, 256, 0, stream>>>(h1, dc2_offw, dc2_offb, off2, C_OUT);
  deform_sample_kernel<<<(BATCH * NK << HW_SHIFT) / 256, 256, 0, stream>>>(h1, off2, col2, C_OUT, K2);
  append_x_kernel<<<((size_t)M_TOTAL * C_IN) / 256, 256, 0, stream>>>(x, col2);
  gemm_bf16_wmma_kernel<<<M_TOTAL / 128, 256, 0, stream>>>(col2, Wt2, out, K2, s2 /*unused*/, bias2, 0);
}
